// AttentionLayer_22024592294002
// MI455X (gfx1250) — matrix operand order repaired and checked by reading, not compiled, not run on hardware
//
#include <hip/hip_runtime.h>
#include <stdint.h>

// ---------------- problem constants ----------------
constexpr int B_   = 2;
constexpr int S_   = 2048;
constexpr int HID_ = 4096;
constexpr int NH_  = 32;
constexpr int NKV_ = 8;
constexpr int D_   = 128;
constexpr int GROUPS_ = NH_ / NKV_;
constexpr int M_   = B_ * S_;          // 4096 rows for all projections

// ---------------- vector types ----------------
typedef __attribute__((ext_vector_type(16))) __bf16 v16bf;
typedef __attribute__((ext_vector_type(8)))  __bf16 v8bf;
typedef __attribute__((ext_vector_type(8)))  float  v8f;
typedef __attribute__((ext_vector_type(4)))  float  v4f;
typedef __attribute__((ext_vector_type(8)))  uint16_t v8u16;

typedef __attribute__((address_space(3))) uint16_t lds_u16;

static __device__ __forceinline__ uint16_t f2bf(float f) {
    uint32_t u = __float_as_uint(f);
    u += 0x7FFFu + ((u >> 16) & 1u);      // round-to-nearest-even
    return (uint16_t)(u >> 16);
}
static __device__ __forceinline__ float bf2f(uint16_t h) {
    return __uint_as_float(((uint32_t)h) << 16);
}
static __device__ __forceinline__ v8bf ld8(const uint16_t* p) {
    return *(const v8bf*)p;               // 16B -> b128 load
}
static __device__ __forceinline__ v16bf cat16(v8bf lo, v8bf hi) {
    v16bf r;
#pragma unroll
    for (int i = 0; i < 8; ++i) { r[i] = lo[i]; r[i + 8] = hi[i]; }
    return r;
}
static __device__ __forceinline__ v8f wmma_bf16(v16bf a, v16bf b, v8f c) {
    // D = A(16x32 bf16) * B(32x16 bf16) + C(16x16 f32)
    return __builtin_amdgcn_wmma_f32_16x16x32_bf16(
        false, a, false, b, (short)0, c, false, false);
}
static __device__ __forceinline__ v8f v8f_zero() {
    v8f z;
#pragma unroll
    for (int i = 0; i < 8; ++i) z[i] = 0.0f;
    return z;
}

// Async 16-byte copy global -> LDS (CDNA5 GLOBAL_LOAD_ASYNC_TO_LDS_B128, ASYNCcnt).
static __device__ __forceinline__ void async_cp_b128(uint16_t* lds_dst, const uint16_t* gsrc) {
    uint32_t loff = (uint32_t)(uintptr_t)(lds_u16*)lds_dst;   // LDS byte offset
    uint64_t ga   = (uint64_t)(uintptr_t)gsrc;
    asm volatile("global_load_async_to_lds_b128 %0, %1, off"
                 :: "v"(loff), "v"(ga) : "memory");
}
static __device__ __forceinline__ void wait_async0() {
    asm volatile("s_wait_asynccnt 0x0" ::: "memory");
}

// ---------------- f32 -> bf16 conversion (8 elems / thread) ----------------
__global__ void cvt_kernel(const float* __restrict__ in, uint16_t* __restrict__ out, long n8) {
    long i = (long)blockIdx.x * blockDim.x + threadIdx.x;
    if (i >= n8) return;
    long base = i * 8;
    v4f a = *(const v4f*)(in + base);
    v4f b = *(const v4f*)(in + base + 4);
    v8u16 r;
#pragma unroll
    for (int j = 0; j < 4; ++j) { r[j] = f2bf(a[j]); r[4 + j] = f2bf(b[j]); }
    *(v8u16*)(out + base) = r;
}

// ---------------- generic C = A * W^T  (bf16 in, f32 accum) ----------------
// A: [M,K] bf16 row-major.  W: [N,K] bf16 row-major (so B[k][n] = W[n][k]).
// Block: 256 threads = 8 waves, tile M=64 x N=256; each wave: 32x64 (2x4 WMMA tiles).
// K-tiles (A: 64x32, B: 256x32) are double-buffered in LDS, filled with
// global_load_async_to_lds_b128 and consumed via ds_load_b128 fragments.
template <bool OUT_BF16>
__global__ void __launch_bounds__(256)
gemm_bt(const uint16_t* __restrict__ A, const uint16_t* __restrict__ W,
        void* __restrict__ Cout, int M, int N, int K)
{
    __shared__ __align__(16) uint16_t sA[2][64 * 32];    //  2 x 4 KB
    __shared__ __align__(16) uint16_t sB[2][256 * 32];   //  2 x 16 KB

    const int tid  = threadIdx.x;
    const int lane = tid & 31;
    const int wave = tid >> 5;
    const int half = lane >> 4;       // 0 or 1 (16-lane halves)
    const int ln   = lane & 15;
    const int wm   = wave & 1;        // 2 waves along M
    const int wn   = wave >> 1;       // 4 waves along N
    const int blockM = blockIdx.y * 64;
    const int blockN = blockIdx.x * 256;

    v8f acc[2][4];
#pragma unroll
    for (int mt = 0; mt < 2; ++mt)
#pragma unroll
        for (int nt = 0; nt < 4; ++nt) acc[mt][nt] = v8f_zero();

    auto stage = [&](int kt, int buf) {
        {   // A tile 64x32: one b128 chunk per thread
            int row = tid >> 2, c4 = (tid & 3) * 8;
            async_cp_b128(&sA[buf][row * 32 + c4],
                          A + (size_t)(blockM + row) * K + kt + c4);
        }
#pragma unroll
        for (int j = 0; j < 4; ++j) {   // B tile 256x32: four b128 chunks per thread
            int idx = tid + 256 * j;
            int row = idx >> 2, c4 = (idx & 3) * 8;
            async_cp_b128(&sB[buf][row * 32 + c4],
                          W + (size_t)(blockN + row) * K + kt + c4);
        }
    };

    stage(0, 0);
    int cur = 0;
    for (int kt = 0; kt < K; kt += 32) {
        wait_async0();          // my slice of buf[cur] landed
        __syncthreads();        // everyone's slice landed; prev buf fully consumed
        if (kt + 32 < K) stage(kt + 32, cur ^ 1);

        // A fragment (16x32): lane<16 holds K 0..15, lane>=16 K 16..31
        v16bf afr[2];
#pragma unroll
        for (int mt = 0; mt < 2; ++mt) {
            const uint16_t* p = &sA[cur][(wm * 32 + mt * 16 + ln) * 32];
            afr[mt] = cat16(ld8(p + half * 16), ld8(p + half * 16 + 8));
        }
        // B fragment (32x16): lane n=ln, K = half*16 .. +15, contiguous in row
        v16bf bfr[4];
#pragma unroll
        for (int nt = 0; nt < 4; ++nt) {
            const uint16_t* p = &sB[cur][(wn * 64 + nt * 16 + ln) * 32];
            bfr[nt] = cat16(ld8(p + half * 16), ld8(p + half * 16 + 8));
        }

#pragma unroll
        for (int mt = 0; mt < 2; ++mt)
#pragma unroll
            for (int nt = 0; nt < 4; ++nt)
                acc[mt][nt] = wmma_bf16(afr[mt], bfr[nt], acc[mt][nt]);

        cur ^= 1;
    }

    // C/D layout: VGPR v -> row v (lanes 0-15) / v+8 (lanes 16-31); col = ln
#pragma unroll
    for (int mt = 0; mt < 2; ++mt)
#pragma unroll
        for (int nt = 0; nt < 4; ++nt)
#pragma unroll
            for (int v = 0; v < 8; ++v) {
                int row = blockM + wm * 32 + mt * 16 + v + 8 * half;
                int col = blockN + wn * 64 + nt * 16 + ln;
                if (OUT_BF16)
                    ((uint16_t*)Cout)[(size_t)row * N + col] = f2bf(acc[mt][nt][v]);
                else
                    ((float*)Cout)[(size_t)row * N + col] = acc[mt][nt][v];
            }
}

// ---------------- RoPE + [B,S,H,D] -> [B,H,S,D] transpose ----------------
__global__ void rope_kernel(const uint16_t* __restrict__ in, uint16_t* __restrict__ out,
                            const int* __restrict__ pos, int H, long total)
{
    long tid = (long)blockIdx.x * blockDim.x + threadIdx.x;
    if (tid >= total) return;
    int d = (int)(tid & 63);                  // 0..63 (half of D)
    int h = (int)((tid >> 6) % H);
    int s = (int)((tid / (64L * H)) % S_);
    int b = (int)(tid / (64L * H * S_));

    const uint16_t* row = in + (size_t)(b * S_ + s) * (H * D_) + h * D_;
    float x0 = bf2f(row[d]);
    float x1 = bf2f(row[d + 64]);
    float p  = (float)pos[s];
    // inv_freq = theta^(-d/64) = exp(-ln(10000) * d / 64)
    float invf = __expf(-9.210340371976184f * (float)d * (1.0f / 64.0f));
    float c, sn;
    __sincosf(p * invf, &sn, &c);
    size_t o = ((size_t)(b * H + h) * S_ + s) * D_ + d;
    out[o]      = f2bf(x0 * c - x1 * sn);
    out[o + 64] = f2bf(x0 * sn + x1 * c);
}

// ---------------- V: [B,S,NKV,D] -> Vt [B,NKV,D,S] ----------------
__global__ void vtrans_kernel(const uint16_t* __restrict__ in, uint16_t* __restrict__ out, long total)
{
    long tid = (long)blockIdx.x * blockDim.x + threadIdx.x;
    if (tid >= total) return;
    int d  = (int)(tid % D_);
    int kv = (int)((tid / D_) % NKV_);
    int s  = (int)((tid / ((long)D_ * NKV_)) % S_);
    int b  = (int)(tid / ((long)D_ * NKV_ * S_));
    out[((size_t)(b * NKV_ + kv) * D_ + d) * S_ + s] =
        in[(size_t)(b * S_ + s) * (NKV_ * D_) + kv * D_ + d];
}

// ---------------- causal flash attention, 1 wave / 16 queries ----------------
// Q: [B,NH,S,D] bf16 (rope'd), K: [B,NKV,S,D] bf16 (rope'd), Vt: [B,NKV,D,S] bf16
// ctx out: [B*S, NH*D] bf16
__global__ void __launch_bounds__(32)
attn_kernel(const uint16_t* __restrict__ Q, const uint16_t* __restrict__ Kr,
            const uint16_t* __restrict__ Vt, uint16_t* __restrict__ ctx)
{
    __shared__ __align__(32) uint16_t lds_p[16 * 32];   // P tile, bf16

    const int lane = threadIdx.x & 31;
    const int half = lane >> 4;
    const int ln   = lane & 15;
    const int qbase = blockIdx.x * 16;
    const int head  = blockIdx.y;             // b*NH + h
    const int b  = head / NH_;
    const int h  = head % NH_;
    const int kv = h / GROUPS_;

    // Preload Q A-fragments for the whole K-loop (16 rows x 128 d)
    const uint16_t* qrow = Q + ((size_t)head * S_ + qbase + ln) * D_;
    v16bf aq[4];
#pragma unroll
    for (int kt = 0; kt < 4; ++kt)
        aq[kt] = cat16(ld8(qrow + kt * 32 + half * 16),
                       ld8(qrow + kt * 32 + half * 16 + 8));

    const uint16_t* kbp = Kr + (size_t)(b * NKV_ + kv) * S_ * D_;
    const uint16_t* vbp = Vt + (size_t)(b * NKV_ + kv) * D_ * S_;

    v8f o[8];
#pragma unroll
    for (int t = 0; t < 8; ++t) o[t] = v8f_zero();
    float mstat[8], lstat[8];
#pragma unroll
    for (int v = 0; v < 8; ++v) { mstat[v] = -3.0e38f; lstat[v] = 0.0f; }

    const float scale = 0.08838834764831845f;   // 1/sqrt(128)
    const int rowb = qbase + 8 * half;          // + v = this lane's row for VGPR v

    for (int kb = 0; kb < qbase + 16; kb += 32) {
        // ---- S = Q K^T for 32 keys (two 16-wide n-tiles) ----
        v8f s0 = v8f_zero(), s1 = v8f_zero();
#pragma unroll
        for (int kt = 0; kt < 4; ++kt) {
            const uint16_t* kp0 = kbp + (size_t)(kb + ln) * D_ + kt * 32 + half * 16;
            const uint16_t* kp1 = kbp + (size_t)(kb + 16 + ln) * D_ + kt * 32 + half * 16;
            v16bf b0 = cat16(ld8(kp0), ld8(kp0 + 8));
            v16bf b1 = cat16(ld8(kp1), ld8(kp1 + 8));
            s0 = wmma_bf16(aq[kt], b0, s0);
            s1 = wmma_bf16(aq[kt], b1, s1);
        }

        // ---- causal mask + online softmax (16-lane row reductions) ----
#pragma unroll
        for (int v = 0; v < 8; ++v) {
            int row = rowb + v;
            float x0 = s0[v] * scale;
            float x1 = s1[v] * scale;
            if (kb + ln > row)      x0 = -1.0e30f;
            if (kb + 16 + ln > row) x1 = -1.0e30f;
            float mx = fmaxf(x0, x1);
#pragma unroll
            for (int msk = 1; msk < 16; msk <<= 1) mx = fmaxf(mx, __shfl_xor(mx, msk, 16));
            float mnew = fmaxf(mstat[v], mx);
            float p0 = __expf(x0 - mnew);
            float p1 = __expf(x1 - mnew);
            float sum = p0 + p1;
#pragma unroll
            for (int msk = 1; msk < 16; msk <<= 1) sum += __shfl_xor(sum, msk, 16);
            float alpha = __expf(mstat[v] - mnew);
            mstat[v] = mnew;
            lstat[v] = lstat[v] * alpha + sum;
#pragma unroll
            for (int t = 0; t < 8; ++t) o[t][v] *= alpha;
            lds_p[(v + 8 * half) * 32 + ln]      = f2bf(p0);
            lds_p[(v + 8 * half) * 32 + 16 + ln] = f2bf(p1);
        }

        // Same-wave LDS ops are in-order; just stop the compiler from reordering.
        __builtin_amdgcn_wave_barrier();
        asm volatile("" ::: "memory");

        // ---- reload P in A-fragment layout, then O += P V ----
        v16bf pa = cat16(ld8(&lds_p[ln * 32 + half * 16]),
                         ld8(&lds_p[ln * 32 + half * 16 + 8]));
        asm volatile("" ::: "memory");
        __builtin_amdgcn_wave_barrier();

#pragma unroll
        for (int t = 0; t < 8; ++t) {
            const uint16_t* vp = vbp + (size_t)(t * 16 + ln) * S_ + kb + half * 16;
            v16bf bv = cat16(ld8(vp), ld8(vp + 8));
            o[t] = wmma_bf16(pa, bv, o[t]);
        }
    }

    // ---- normalize + write ctx [B*S, NH*D] ----
#pragma unroll
    for (int v = 0; v < 8; ++v) {
        float inv = 1.0f / lstat[v];
        int srow = qbase + v + 8 * half;
        size_t base = ((size_t)(b * S_ + srow) * NH_ + h) * D_;
#pragma unroll
        for (int t = 0; t < 8; ++t)
            ctx[base + t * 16 + ln] = f2bf(o[t][v] * inv);
    }
}

// ---------------- host-side orchestration ----------------
extern "C" void kernel_launch(void* const* d_in, const int* in_sizes, int n_in,
                              void* d_out, int out_size, void* d_ws, size_t ws_size,
                              hipStream_t stream)
{
    (void)in_sizes; (void)n_in; (void)out_size; (void)ws_size;

    const float* x   = (const float*)d_in[0];
    const int*   pos = (const int*)d_in[1];
    const float* wq  = (const float*)d_in[2];
    const float* wk  = (const float*)d_in[3];
    const float* wv  = (const float*)d_in[4];
    const float* wo  = (const float*)d_in[5];
    float* out = (float*)d_out;

    uint16_t* ws = (uint16_t*)d_ws;
    size_t off = 0;
    uint16_t* xb   = ws + off; off += (size_t)M_ * HID_;
    uint16_t* wqb  = ws + off; off += (size_t)NH_ * D_ * HID_;
    uint16_t* wkb  = ws + off; off += (size_t)NKV_ * D_ * HID_;
    uint16_t* wvb  = ws + off; off += (size_t)NKV_ * D_ * HID_;
    uint16_t* wob  = ws + off; off += (size_t)HID_ * NH_ * D_;
    uint16_t* qlin = ws + off; off += (size_t)M_ * NH_ * D_;     // reused as ctx
    uint16_t* klin = ws + off; off += (size_t)M_ * NKV_ * D_;
    uint16_t* vlin = ws + off; off += (size_t)M_ * NKV_ * D_;
    uint16_t* qr   = ws + off; off += (size_t)B_ * NH_ * S_ * D_;
    uint16_t* kr   = ws + off; off += (size_t)B_ * NKV_ * S_ * D_;
    uint16_t* vt   = ws + off; off += (size_t)B_ * NKV_ * S_ * D_;
    uint16_t* ctx  = qlin;   // qlin dead after rope_kernel(Q)

    auto cvt = [&](const float* src, uint16_t* dst, size_t n) {
        long n8 = (long)(n / 8);
        cvt_kernel<<<dim3((unsigned)((n8 + 255) / 256)), 256, 0, stream>>>(src, dst, n8);
    };
    cvt(x,  xb,  (size_t)M_ * HID_);
    cvt(wq, wqb, (size_t)NH_ * D_ * HID_);
    cvt(wk, wkb, (size_t)NKV_ * D_ * HID_);
    cvt(wv, wvb, (size_t)NKV_ * D_ * HID_);
    cvt(wo, wob, (size_t)HID_ * NH_ * D_);

    // QKV projections (bf16 WMMA, f32 accumulate, bf16 out)
    gemm_bt<true><<<dim3((NH_ * D_) / 256, M_ / 64), 256, 0, stream>>>(xb, wqb, qlin, M_, NH_ * D_, HID_);
    gemm_bt<true><<<dim3((NKV_ * D_) / 256, M_ / 64), 256, 0, stream>>>(xb, wkb, klin, M_, NKV_ * D_, HID_);
    gemm_bt<true><<<dim3((NKV_ * D_) / 256, M_ / 64), 256, 0, stream>>>(xb, wvb, vlin, M_, NKV_ * D_, HID_);

    // RoPE + transposes
    { long n = (long)B_ * S_ * NH_ * 64;
      rope_kernel<<<dim3((unsigned)((n + 255) / 256)), 256, 0, stream>>>(qlin, qr, pos, NH_, n); }
    { long n = (long)B_ * S_ * NKV_ * 64;
      rope_kernel<<<dim3((unsigned)((n + 255) / 256)), 256, 0, stream>>>(klin, kr, pos, NKV_, n); }
    { long n = (long)B_ * S_ * NKV_ * D_;
      vtrans_kernel<<<dim3((unsigned)((n + 255) / 256)), 256, 0, stream>>>(vlin, vt, n); }

    // Causal flash attention -> ctx
    attn_kernel<<<dim3(S_ / 16, B_ * NH_), 32, 0, stream>>>(qr, kr, vt, ctx);

    // Output projection: out = ctx @ wo^T  (f32 out)
    gemm_bt<false><<<dim3(HID_ / 256, M_ / 64), 256, 0, stream>>>(ctx, wob, out, M_, HID_, NH_ * D_);
}
